// GPSLayer_21088289423717
// MI455X (gfx1250) — matrix operand order repaired and checked by reading, not compiled, not run on hardware
//
#include <hip/hip_runtime.h>
#include <hip/hip_bf16.h>
#include <stdint.h>

// ---------------- constants for this problem ----------------
#define NN    4096
#define EE    262144
#define DN    256
#define DE    128
#define DIN   384          // DN + DE
#define NH    4
#define HD    64
#define DQKV  768
#define DFF   1024

typedef __attribute__((ext_vector_type(16))) __bf16 v16bf;
typedef __attribute__((ext_vector_type(8)))  float  v8f;
typedef int v4i_async __attribute__((vector_size(16)));

// ---------------- helpers ----------------
__device__ __forceinline__ __bf16 f2bf(float f) {
    unsigned u = __builtin_bit_cast(unsigned, f);
    unsigned r = (u + 0x7FFFu + ((u >> 16) & 1u)) >> 16;
    unsigned short h = (unsigned short)r;
    return __builtin_bit_cast(__bf16, h);
}
__device__ __forceinline__ unsigned pack2bf(float a, float b) {
    unsigned lo = (unsigned)__builtin_bit_cast(unsigned short, f2bf(a));
    unsigned hi = (unsigned)__builtin_bit_cast(unsigned short, f2bf(b));
    return lo | (hi << 16);
}
__device__ __forceinline__ float gelu_exact(float v) {
    return 0.5f * v * (1.0f + erff(v * 0.70710678118654752440f));
}
// A-fragment (16x32 bf16): slot j -> K index = 16*half + j, given lane-half (0: lanes 0-15, 1: lanes 16-31)
__device__ __forceinline__ int a_kidx(int j, int khalf) {
    return 16 * khalf + j;
}
// B-fragment (32x16 bf16): slot j -> K index = 16*half + j
__device__ __forceinline__ int b_kidx(int j, int khalf) { return 16 * khalf + j; }

// guaranteed native f32 atomic add (no CAS loop)
__device__ __forceinline__ void atom_add_f32(float* p, float v) {
    unsigned long long a = (unsigned long long)(uintptr_t)p;
    asm volatile("global_atomic_add_f32 %0, %1, off" : : "v"(a), "v"(v) : "memory");
}

// 16-byte global -> LDS copy; prefer the CDNA5 async-to-LDS path (ASYNCcnt).
#if defined(__gfx1250__) && __has_builtin(__builtin_amdgcn_global_load_async_to_lds_b128)
#define HAVE_ASYNC_LDS 1
#else
#define HAVE_ASYNC_LDS 0
#endif

__device__ __forceinline__ void copy16_g2l(const void* g, void* l) {
#if HAVE_ASYNC_LDS
    __builtin_amdgcn_global_load_async_to_lds_b128(
        (__attribute__((address_space(1))) v4i_async*)g,
        (__attribute__((address_space(3))) v4i_async*)l, 0, 0);
#else
    *(uint4*)l = *(const uint4*)g;
#endif
}
__device__ __forceinline__ void wait_g2l() {
#if HAVE_ASYNC_LDS
#if __has_builtin(__builtin_amdgcn_s_wait_asynccnt)
    __builtin_amdgcn_s_wait_asynccnt(0);
#else
    asm volatile("s_wait_asynccnt 0x0" ::: "memory");
#endif
#endif
}

// ---------------- tiny elementwise kernels ----------------
__global__ void cvt_f32_bf16(const float* __restrict__ in, __bf16* __restrict__ out, int n) {
    int i = blockIdx.x * blockDim.x + threadIdx.x;
    if (i < n) out[i] = f2bf(in[i]);
}
__global__ void zero_f32(float* __restrict__ p, int n) {
    int i = blockIdx.x * blockDim.x + threadIdx.x;
    if (i < n) p[i] = 0.0f;
}
// pre = bf16((1+eps)*x + agg)
__global__ void pre_kernel(const float* __restrict__ x, const float* __restrict__ agg,
                           const float* __restrict__ eps, __bf16* __restrict__ out, int n) {
    int i = blockIdx.x * blockDim.x + threadIdx.x;
    if (i < n) out[i] = f2bf((1.0f + eps[0]) * x[i] + agg[i]);
}

// convert f32 weight [K,N] into WMMA-B-fragment-major bf16:
// out[((ct*KT + kt)*32 + lane)*16 + j] = W[kt*32 + 16*(lane>>4) + j][ct*16 + (lane&15)]
__global__ void cvt_pack_b(const float* __restrict__ W, __bf16* __restrict__ out, int K, int N) {
    int i = blockIdx.x * blockDim.x + threadIdx.x;
    if (i >= K * N) return;
    int j    = i & 15;
    int lane = (i >> 4) & 31;
    int t    = i >> 9;                  // fragment id
    int KT   = K >> 5;
    int kt   = t % KT, ct = t / KT;
    int half = lane >> 4, ln15 = lane & 15;
    int k = kt * 32 + 16 * half + j;
    int n = ct * 16 + ln15;
    out[i] = f2bf(W[k * N + n]);
}

// repack runtime qkv (bf16 [N,768]) into fragment-major Q (A-layout), K^T and V (B-layout)
__global__ void pack_q_kernel(const __bf16* __restrict__ qkv, __bf16* __restrict__ Qp) {
    int i = blockIdx.x * blockDim.x + threadIdx.x;      // NH*256*2*512 = 1M
    int j = i & 15, lane = (i >> 4) & 31, ktf = (i >> 9) & 1;
    int rt = (i >> 10) & 255, head = (i >> 18) & 3;
    int half = lane >> 4, ln15 = lane & 15;
    Qp[i] = qkv[(rt * 16 + ln15) * DQKV + head * HD + ktf * 32 + a_kidx(j, half)];
}
__global__ void pack_k_kernel(const __bf16* __restrict__ qkv, __bf16* __restrict__ Kp) {
    int i = blockIdx.x * blockDim.x + threadIdx.x;      // NH*256*2*512 = 1M
    int j = i & 15, lane = (i >> 4) & 31, ktf = (i >> 9) & 1;
    int kt16 = (i >> 10) & 255, head = (i >> 18) & 3;
    int half = lane >> 4, ln15 = lane & 15;
    Kp[i] = qkv[(kt16 * 16 + ln15) * DQKV + DN + head * HD + ktf * 32 + b_kidx(j, half)];
}
__global__ void pack_v_kernel(const __bf16* __restrict__ qkv, __bf16* __restrict__ Vp) {
    int i = blockIdx.x * blockDim.x + threadIdx.x;      // NH*128*4*512 = 1M
    int j = i & 15, lane = (i >> 4) & 31, ct = (i >> 9) & 3;
    int kb32 = (i >> 11) & 127, head = (i >> 18) & 3;
    int half = lane >> 4, ln15 = lane & 15;
    int row = kb32 * 32 + 16 * half + j;                // key index (K dim of P@V)
    Vp[i] = qkv[row * DQKV + 2 * DN + head * HD + ct * 16 + ln15];
}

// ---------------- LayerNorm: out = LN(a + b)  (one row per block, D=256) ----------------
__global__ __launch_bounds__(256) void ln_kernel(const float* __restrict__ a,
                                                 const float* __restrict__ b,
                                                 const float* __restrict__ g,
                                                 const float* __restrict__ be,
                                                 float* __restrict__ of,
                                                 __bf16* __restrict__ ob) {
    __shared__ float red[256];
    int row = blockIdx.x, t = threadIdx.x;
    float x = a[row * DN + t] + b[row * DN + t];
    red[t] = x; __syncthreads();
    for (int s = 128; s > 0; s >>= 1) { if (t < s) red[t] += red[t + s]; __syncthreads(); }
    float mu = red[0] * (1.0f / DN);
    __syncthreads();
    float d = x - mu;
    red[t] = d * d; __syncthreads();
    for (int s = 128; s > 0; s >>= 1) { if (t < s) red[t] += red[t + s]; __syncthreads(); }
    float var = red[0] * (1.0f / DN);
    float y = d * rsqrtf(var + 1e-5f) * g[t] + be[t];
    of[row * DN + t] = y;
    if (ob) ob[row * DN + t] = f2bf(y);
}

// ---------------- generic bf16 WMMA GEMM: C = act(A[M,K] @ Bp + bias) ----------------
// Bp is fragment-major (see cvt_pack_b). block = 256 threads (8 waves);
// one 16-row tile per block; waves stride over 16-col tiles.
__global__ __launch_bounds__(256) void gemm_bf16(const __bf16* __restrict__ A,
                                                 const __bf16* __restrict__ Bp,
                                                 const float* __restrict__ bias,
                                                 float* __restrict__ Cf,
                                                 __bf16* __restrict__ Cb,
                                                 int K, int N, int act) {
    extern __shared__ __align__(16) __bf16 As[];   // 16 * K bf16
    int rowbase = blockIdx.x * 16;
    {   // stage A tile via async-to-LDS (16B per lane per issue)
        const __bf16* Arow = A + (size_t)rowbase * K;
        int chunks = (16 * K) >> 3;                // 8 bf16 per 16B chunk
        for (int c = threadIdx.x; c < chunks; c += 256)
            copy16_g2l(Arow + c * 8, As + c * 8);
    }
    wait_g2l();
    __syncthreads();

    int wave = threadIdx.x >> 5, lane = threadIdx.x & 31;
    int half = lane >> 4, ln15 = lane & 15;
    int KT = K >> 5, ntiles = N >> 4;
    const v16bf* Bv = (const v16bf*)Bp;
    for (int ct = wave; ct < ntiles; ct += 8) {
        int nb = ct * 16;
        v8f acc = {};
        for (int kt = 0; kt < KT; ++kt) {
            v16bf a;
#pragma unroll
            for (int j = 0; j < 16; ++j)
                a[j] = As[ln15 * K + kt * 32 + a_kidx(j, half)];
            v16bf b = Bv[(ct * KT + kt) * 32 + lane];
            acc = __builtin_amdgcn_wmma_f32_16x16x32_bf16(false, a, false, b,
                                                          (short)0, acc, false, false);
        }
        float bs = bias[nb + ln15];
#pragma unroll
        for (int r = 0; r < 8; ++r) {
            float v = acc[r] + bs;
            if (act == 1) v = gelu_exact(v);
            int row = rowbase + r + 8 * half;
            int col = nb + ln15;
            if (Cf) Cf[row * N + col] = v;
            if (Cb) Cb[row * N + col] = f2bf(v);
        }
    }
}

// ---------------- edge MLP: gather(bf16 x) -> WMMA -> GELU -> atomic scatter-sum ----------------
__global__ __launch_bounds__(256) void msg_kernel(const __bf16* __restrict__ xb,  // [N,256] bf16
                                                  const float* __restrict__ ef,
                                                  const long long* __restrict__ eidx,
                                                  const __bf16* __restrict__ Wp,  // packed [384,256]
                                                  const float* __restrict__ bias, // [256]
                                                  float* __restrict__ agg) {      // [N,256]
    __shared__ __align__(16) __bf16 As[16 * DIN];
    __shared__ int src_s[16];
    __shared__ int tgt_s[16];
    int e0 = blockIdx.x * 16;
    if (threadIdx.x < 16) {
        src_s[threadIdx.x] = (int)eidx[e0 + threadIdx.x];
        tgt_s[threadIdx.x] = (int)eidx[EE + e0 + threadIdx.x];
    }
    __syncthreads();

    // node part: 16 rows x 256 bf16, gathered as 16B chunks (async-to-LDS)
    {
        int c = threadIdx.x;                       // 512 chunks total, 2 issues/lane
#pragma unroll
        for (int it = 0; it < 2; ++it, c += 256) {
            int m = c >> 5, sub = c & 31;
            copy16_g2l(xb + (size_t)src_s[m] * DN + sub * 8,
                       As + m * DIN + sub * 8);
        }
    }
    // edge part: 16 rows x 128 f32, convert in-register, packed b32 LDS stores
    {
        int m = threadIdx.x >> 4, t16 = threadIdx.x & 15;
        const float4* ep = (const float4*)(ef + (size_t)(e0 + m) * DE + t16 * 8);
        float4 f0 = ep[0], f1 = ep[1];
        unsigned* Au = (unsigned*)As;
        int ui = (m * DIN + DN + t16 * 8) >> 1;
        Au[ui + 0] = pack2bf(f0.x, f0.y);
        Au[ui + 1] = pack2bf(f0.z, f0.w);
        Au[ui + 2] = pack2bf(f1.x, f1.y);
        Au[ui + 3] = pack2bf(f1.z, f1.w);
    }
    wait_g2l();
    __syncthreads();

    int wave = threadIdx.x >> 5, lane = threadIdx.x & 31;
    int half = lane >> 4, ln15 = lane & 15;
    const v16bf* Wv = (const v16bf*)Wp;
    const int KT = DIN / 32;                       // 12
#pragma unroll
    for (int i = 0; i < 2; ++i) {
        int ct = wave * 2 + i;
        int nb = ct * 16;
        v8f acc = {};
        for (int kt = 0; kt < KT; ++kt) {
            v16bf a;
#pragma unroll
            for (int j = 0; j < 16; ++j)
                a[j] = As[ln15 * DIN + kt * 32 + a_kidx(j, half)];
            v16bf b = Wv[(ct * KT + kt) * 32 + lane];
            acc = __builtin_amdgcn_wmma_f32_16x16x32_bf16(false, a, false, b,
                                                          (short)0, acc, false, false);
        }
        float bs = bias[nb + ln15];
#pragma unroll
        for (int r = 0; r < 8; ++r) {
            float v = gelu_exact(acc[r] + bs);
            int m = r + 8 * half;
            atom_add_f32(&agg[tgt_s[m] * DN + nb + ln15], v);
        }
    }
}

// ---------------- flash attention: one wave per (head, 16-query tile) ----------------
// Qp/Kp/Vp are fragment-major (pack_*_kernel); all inner-loop operands are vector loads.
__global__ __launch_bounds__(256) void attn_kernel(const __bf16* __restrict__ Qp,
                                                   const __bf16* __restrict__ Kp,
                                                   const __bf16* __restrict__ Vp,
                                                   __bf16* __restrict__ outb) {  // [N,256] bf16
    __shared__ __bf16 Pws[8][16 * 32];    // per-wave P staging (D-layout -> A-layout)
    int wave = threadIdx.x >> 5, lane = threadIdx.x & 31;
    int half = lane >> 4, ln15 = lane & 15;
    int head = blockIdx.y;
    int rt = blockIdx.x * 8 + wave;
    int r0 = rt * 16;
    const v16bf* Qv = (const v16bf*)Qp;
    const v16bf* Kv = (const v16bf*)Kp;
    const v16bf* Vv = (const v16bf*)Vp;

    // Q fragments (held for the whole key sweep)
    v16bf aq[2];
#pragma unroll
    for (int ktf = 0; ktf < 2; ++ktf)
        aq[ktf] = Qv[((head * 256 + rt) * 2 + ktf) * 32 + lane];

    v8f acc[4] = {};                       // O accumulators (4 col tiles of 16 = HD)
    float mrun[8], lrun[8];
#pragma unroll
    for (int r = 0; r < 8; ++r) { mrun[r] = -1e30f; lrun[r] = 0.0f; }

    for (int kb = 0; kb < NN; kb += 32) {
        // scores for 32 keys: two 16x16 tiles
        v8f s[2];
#pragma unroll
        for (int st = 0; st < 2; ++st) {
            int kt16 = (kb >> 4) + st;
            v8f sa = {};
#pragma unroll
            for (int ktf = 0; ktf < 2; ++ktf) {
                v16bf bk = Kv[((head * 256 + kt16) * 2 + ktf) * 32 + lane];
                sa = __builtin_amdgcn_wmma_f32_16x16x32_bf16(false, aq[ktf], false, bk,
                                                             (short)0, sa, false, false);
            }
            s[st] = sa;
        }
        // online softmax per row (rows live in VGPR index, cols in lanes of a half)
        float p0[8], p1[8];
#pragma unroll
        for (int r = 0; r < 8; ++r) {
            float s0 = s[0][r] * 0.125f, s1 = s[1][r] * 0.125f;   // HD^-0.5 = 1/8
            float mx = fmaxf(s0, s1);
            for (int off = 8; off >= 1; off >>= 1) mx = fmaxf(mx, __shfl_xor(mx, off, 32));
            float mnew = fmaxf(mrun[r], mx);
            float sc = __expf(mrun[r] - mnew);          // args <= 0: native exp is safe
            float e0v = __expf(s0 - mnew), e1v = __expf(s1 - mnew);
            float rs = e0v + e1v;
            for (int off = 8; off >= 1; off >>= 1) rs += __shfl_xor(rs, off, 32);
            lrun[r] = lrun[r] * sc + rs;
            mrun[r] = mnew;
#pragma unroll
            for (int ct = 0; ct < 4; ++ct) acc[ct][r] *= sc;
            p0[r] = e0v; p1[r] = e1v;
        }
        // stage P through wave-private LDS to re-layout into an A fragment
#pragma unroll
        for (int r = 0; r < 8; ++r) {
            Pws[wave][(r + 8 * half) * 32 + ln15]      = f2bf(p0[r]);
            Pws[wave][(r + 8 * half) * 32 + 16 + ln15] = f2bf(p1[r]);
        }
        v16bf ap;
#pragma unroll
        for (int j = 0; j < 16; ++j)
            ap[j] = Pws[wave][ln15 * 32 + a_kidx(j, half)];
        // O += P @ V
        int kb32 = kb >> 5;
#pragma unroll
        for (int ct = 0; ct < 4; ++ct) {
            v16bf bv = Vv[((head * 128 + kb32) * 4 + ct) * 32 + lane];
            acc[ct] = __builtin_amdgcn_wmma_f32_16x16x32_bf16(false, ap, false, bv,
                                                              (short)0, acc[ct], false, false);
        }
    }
#pragma unroll
    for (int r = 0; r < 8; ++r) {
        float inv = 1.0f / lrun[r];
        int row = r0 + r + 8 * half;
#pragma unroll
        for (int ct = 0; ct < 4; ++ct)
            outb[row * DN + head * HD + ct * 16 + ln15] = f2bf(acc[ct][r] * inv);
    }
}

// ---------------- host orchestration ----------------
extern "C" void kernel_launch(void* const* d_in, const int* in_sizes, int n_in,
                              void* d_out, int out_size, void* d_ws, size_t ws_size,
                              hipStream_t stream) {
    const float*     x     = (const float*)d_in[0];
    const float*     ef    = (const float*)d_in[1];
    const long long* eidx  = (const long long*)d_in[2];
    const float*     eps   = (const float*)d_in[3];
    const float*     msg_w = (const float*)d_in[4];
    const float*     msg_b = (const float*)d_in[5];
    const float*     upd_w = (const float*)d_in[6];
    const float*     upd_b = (const float*)d_in[7];
    const float*     qkv_w = (const float*)d_in[8];
    const float*     qkv_b = (const float*)d_in[9];
    const float*     out_w = (const float*)d_in[10];
    const float*     out_b = (const float*)d_in[11];
    const float*     f1_w  = (const float*)d_in[12];
    const float*     f1_b  = (const float*)d_in[13];
    const float*     f2_w  = (const float*)d_in[14];
    const float*     f2_b  = (const float*)d_in[15];
    const float*     ln1g  = (const float*)d_in[16];
    const float*     ln1b  = (const float*)d_in[17];
    const float*     ln2g  = (const float*)d_in[18];
    const float*     ln2b  = (const float*)d_in[19];
    const float*     ln3g  = (const float*)d_in[20];
    const float*     ln3b  = (const float*)d_in[21];

    char* base = (char*)d_ws;
    size_t off = 0;
    auto alloc = [&](size_t bytes) { char* p = base + off; off += (bytes + 255) & ~(size_t)255; return p; };

    __bf16* msgWb = (__bf16*)alloc((size_t)DIN * DN * 2);
    __bf16* updWb = (__bf16*)alloc((size_t)DN * DN * 2);
    __bf16* qkvWb = (__bf16*)alloc((size_t)DN * DQKV * 2);
    __bf16* outWb = (__bf16*)alloc((size_t)DN * DN * 2);
    __bf16* f1Wb  = (__bf16*)alloc((size_t)DN * DFF * 2);
    __bf16* f2Wb  = (__bf16*)alloc((size_t)DFF * DN * 2);
    __bf16* xb    = (__bf16*)alloc((size_t)NN * DN * 2);
    float*  agg   = (float*) alloc((size_t)NN * DN * 4);
    __bf16* preb  = (__bf16*)alloc((size_t)NN * DN * 2);
    float*  mpnn  = (float*) alloc((size_t)NN * DN * 4);
    float*  h1    = (float*) alloc((size_t)NN * DN * 4);
    __bf16* h1b   = (__bf16*)alloc((size_t)NN * DN * 2);
    __bf16* qkvb  = (__bf16*)alloc((size_t)NN * DQKV * 2);
    __bf16* Qp    = (__bf16*)alloc((size_t)NH * 256 * 2 * 512 * 2);
    __bf16* Kp    = (__bf16*)alloc((size_t)NH * 256 * 2 * 512 * 2);
    __bf16* Vp    = (__bf16*)alloc((size_t)NH * 128 * 4 * 512 * 2);
    __bf16* attnb = (__bf16*)alloc((size_t)NN * DN * 2);
    float*  oproj = (float*) alloc((size_t)NN * DN * 4);
    float*  h2    = (float*) alloc((size_t)NN * DN * 4);
    __bf16* h2b   = (__bf16*)alloc((size_t)NN * DN * 2);
    __bf16* t1b   = (__bf16*)alloc((size_t)NN * DFF * 2);
    float*  ffn2  = (float*) alloc((size_t)NN * DN * 4);

    auto cvtp = [&](const float* src, __bf16* dst, int K, int N) {
        cvt_pack_b<<<(K * N + 255) / 256, 256, 0, stream>>>(src, dst, K, N);
    };
    cvtp(msg_w, msgWb, DIN, DN);
    cvtp(upd_w, updWb, DN, DN);
    cvtp(qkv_w, qkvWb, DN, DQKV);
    cvtp(out_w, outWb, DN, DN);
    cvtp(f1_w,  f1Wb,  DN, DFF);
    cvtp(f2_w,  f2Wb,  DFF, DN);
    cvt_f32_bf16<<<(NN * DN + 255) / 256, 256, 0, stream>>>(x, xb, NN * DN);

    zero_f32<<<(NN * DN + 255) / 256, 256, 0, stream>>>(agg, NN * DN);

    // --- MPNN message + scatter sum ---
    msg_kernel<<<EE / 16, 256, 0, stream>>>(xb, ef, eidx, msgWb, msg_b, agg);

    // --- update MLP + LN1 ---
    pre_kernel<<<(NN * DN + 255) / 256, 256, 0, stream>>>(x, agg, eps, preb, NN * DN);
    gemm_bf16<<<NN / 16, 256, 16 * DN * 2, stream>>>(preb, updWb, upd_b, mpnn, nullptr, DN, DN, 1);
    ln_kernel<<<NN, 256, 0, stream>>>(x, mpnn, ln1g, ln1b, h1, h1b);

    // --- attention (bf16 end-to-end, fragment-major K/V) ---
    gemm_bf16<<<NN / 16, 256, 16 * DN * 2, stream>>>(h1b, qkvWb, qkv_b, nullptr, qkvb, DN, DQKV, 0);
    {
        int npk = NH * 256 * 2 * 512;
        pack_q_kernel<<<npk / 256, 256, 0, stream>>>(qkvb, Qp);
        pack_k_kernel<<<npk / 256, 256, 0, stream>>>(qkvb, Kp);
        pack_v_kernel<<<npk / 256, 256, 0, stream>>>(qkvb, Vp);
    }
    dim3 agrid(NN / 16 / 8, NH);
    attn_kernel<<<agrid, 256, 0, stream>>>(Qp, Kp, Vp, attnb);
    gemm_bf16<<<NN / 16, 256, 16 * DN * 2, stream>>>(attnb, outWb, out_b, oproj, nullptr, DN, DN, 0);
    ln_kernel<<<NN, 256, 0, stream>>>(h1, oproj, ln2g, ln2b, h2, h2b);

    // --- FFN + LN3 ---
    gemm_bf16<<<NN / 16, 256, 16 * DN * 2, stream>>>(h2b, f1Wb, f1_b, nullptr, t1b, DN, DFF, 1);
    gemm_bf16<<<NN / 16, 256, 16 * DFF * 2, stream>>>(t1b, f2Wb, f2_b, ffn2, nullptr, DFF, DN, 0);
    ln_kernel<<<NN, 256, 0, stream>>>(h2, ffn2, ln3g, ln3b, (float*)d_out, nullptr);
}
